// multimodal_attention_layer_type3_91293824844415
// MI455X (gfx1250) — matrix operand order repaired and checked by reading, not compiled, not run on hardware
//
#include <hip/hip_runtime.h>
#include <cstdint>
#include <cstddef>

#define L_ 30000
#define W_ 300
#define SHIFT_ 150
#define H_ 4
#define NW_ 100
#define NL_ 6
#define C_ 128
#define B_ 2
#define WP_ 320     // window padded to multiple of 32
#define HD_ 32

typedef __attribute__((ext_vector_type(16))) _Float16 v16h;
typedef __attribute__((ext_vector_type(8)))  float    v8f;
typedef __attribute__((ext_vector_type(8)))  _Float16 h8v;
typedef __attribute__((ext_vector_type(4)))  _Float16 h4v;

__device__ __forceinline__ v8f wmma_f32_f16(v16h a, v16h b, v8f c) {
  // D = A(16x32 f16) * B(32x16 f16) + C(16x16 f32)
  return __builtin_amdgcn_wmma_f32_16x16x32_f16(false, a, false, b, (short)0, c, false, false);
}

// A fragment: 16x32 (MxK) from row-major f16, leading dim ld (rows 16B-aligned).
// lanes 0-15: row=lane, K={0..7,16..23}; lanes 16-31: row=lane-16, K={8..15,24..31}
__device__ __forceinline__ v16h load_a_frag(const _Float16* p, int ld) {
  const int lane = threadIdx.x & 31;
  const int row  = lane & 15;
  const int kb   = (lane >> 4) << 3;      // 0 or 8
  const _Float16* r = p + (size_t)row * ld + kb;
  h8v lo = *(const h8v*)r;
  h8v hi = *(const h8v*)(r + 16);
  v16h a;
#pragma unroll
  for (int j = 0; j < 8; ++j) { a[j] = lo[j]; a[8 + j] = hi[j]; }
  return a;
}

// B fragment from B^T layout [n][k] (column n contiguous over k), ld = k-stride.
__device__ __forceinline__ v16h load_bt_frag(const _Float16* p, int ld) {
  const int lane = threadIdx.x & 31;
  const int col  = lane & 15;
  const int kb   = (lane >> 4) << 3;      // 0 or 8
  const _Float16* r = p + (size_t)col * ld + kb;
  h8v lo = *(const h8v*)r;
  h8v hi = *(const h8v*)(r + 16);
  v16h b;
#pragma unroll
  for (int j = 0; j < 16; ++j) b[j] = (j < 8) ? lo[j] : hi[j - 8];
  return b;
}

// gfx1250 async global->LDS copy (ASYNCcnt) -----------------------------------
__device__ __forceinline__ void async_copy16(unsigned lds_byte_off, const void* g) {
  asm volatile("global_load_async_to_lds_b128 %0, %1, off"
               :: "v"(lds_byte_off), "v"(g) : "memory");
}
__device__ __forceinline__ void wait_async0() {
  asm volatile("s_wait_asynccnt 0x0" ::: "memory");
}

// ---------------- prep kernels ----------------

// src: [NL][K][N] f32 -> dst: [NL][N][K] f16
__global__ void wcast_kernel(const float* __restrict__ src, _Float16* __restrict__ dst, int K, int N) {
  long total = (long)NL_ * K * N;
  for (long idx = (long)blockIdx.x * blockDim.x + threadIdx.x; idx < total;
       idx += (long)gridDim.x * blockDim.x) {
    long layer = idx / ((long)K * N);
    long rem   = idx % ((long)K * N);
    int k = (int)(rem / N), n = (int)(rem % N);
    dst[layer * (long)K * N + (long)n * K + k] = (_Float16)src[idx];
  }
}

__global__ void init_kernel(const float* __restrict__ s1, const float* __restrict__ s2,
                            float* __restrict__ x) {
  size_t idx = (size_t)blockIdx.x * blockDim.x + threadIdx.x;
  if (idx >= (size_t)B_ * C_ * L_) return;
  int l = (int)(idx % L_);
  int c = (int)((idx / L_) % C_);
  int b = (int)(idx / ((size_t)C_ * L_));
  x[((size_t)b * L_ + l) * C_ + c] = s1[idx] * s2[idx];
}

__global__ void pack_kernel(const float* __restrict__ x, float* __restrict__ out) {
  size_t idx = (size_t)blockIdx.x * blockDim.x + threadIdx.x;
  if (idx >= (size_t)B_ * C_ * L_) return;
  int l = (int)(idx % L_);
  int c = (int)((idx / L_) % C_);
  int b = (int)(idx / ((size_t)C_ * L_));
  out[idx] = x[((size_t)b * L_ + l) * C_ + c];
}

// ---------------- layernorm (+ optional roll by -shift), wave32 shuffles ------
__global__ void __launch_bounds__(128)
ln_kernel(const float* __restrict__ x, _Float16* __restrict__ h,
          const float* __restrict__ g, const float* __restrict__ bt, int shift) {
  __shared__ float p1[4], p2[4];
  int row = blockIdx.x;                 // 0 .. B*L-1
  int b = row / L_, l = row % L_;
  int lsrc = l + shift; if (lsrc >= L_) lsrc -= L_;
  int t = threadIdx.x;
  float val = x[((size_t)b * L_ + lsrc) * C_ + t];
  float s1v = val, s2v = val * val;
#pragma unroll
  for (int off = 16; off; off >>= 1) {
    s1v += __shfl_xor(s1v, off);
    s2v += __shfl_xor(s2v, off);
  }
  int wave = t >> 5;
  if ((t & 31) == 0) { p1[wave] = s1v; p2[wave] = s2v; }
  __syncthreads();
  float tot1 = p1[0] + p1[1] + p1[2] + p1[3];
  float tot2 = p2[0] + p2[1] + p2[2] + p2[3];
  float m   = tot1 * (1.0f / C_);
  float inv = rsqrtf(tot2 * (1.0f / C_) - m * m + 1e-5f);
  h[((size_t)b * L_ + l) * C_ + t] = (_Float16)((val - m) * inv * g[t] + bt[t]);
}

// ---------------- QKV GEMM: wave does 1 m-tile x 6 n-tiles (A reuse) ----------
__global__ void __launch_bounds__(256)
qkv_gemm_kernel(const _Float16* __restrict__ h, const _Float16* __restrict__ wt,
                const float* __restrict__ bias,
                _Float16* __restrict__ q, _Float16* __restrict__ k, _Float16* __restrict__ v) {
  const int MT = (B_ * L_) / 16;        // 3750
  int wid = blockIdx.x * 8 + (threadIdx.x >> 5);
  if (wid >= MT * 4) return;
  int mt = wid >> 2, ng = wid & 3;      // ng: group of 6 n-tiles
  int m0 = mt * 16;
  const int lane = threadIdx.x & 31, col = lane & 15, hi = lane >> 4;

  v16h af[4];
#pragma unroll
  for (int kt = 0; kt < 4; ++kt) af[kt] = load_a_frag(h + (size_t)m0 * C_ + kt * 32, C_);

  size_t pre[8];
#pragma unroll
  for (int r = 0; r < 8; ++r) {
    int m = m0 + r + hi * 8;
    int bb = m / L_, l = m - bb * L_;
    int nw = l / W_, w = l - nw * W_;
    pre[r] = ((size_t)bb * NW_ + nw) * ((size_t)H_ * W_ * HD_) + (size_t)w * HD_;
  }

  const float scale = 0.17677669529663687f;   // hd^-0.5
  for (int j = 0; j < 6; ++j) {
    int nt = ng * 6 + j, n0 = nt * 16;
    v8f c = {};
#pragma unroll
    for (int kt = 0; kt < 4; ++kt)
      c = wmma_f32_f16(af[kt], load_bt_frag(wt + (size_t)n0 * C_ + kt * 32, C_), c);
    int n = n0 + col;
    int part = n >> 7, cc = n & 127, hh = cc >> 5, d = cc & 31;
    float bv = bias[n];
    float sc = (part == 0) ? scale : 1.0f;
    _Float16* dst = (part == 0) ? q : (part == 1 ? k : v);
    size_t hoff = (size_t)hh * (W_ * HD_) + d;
#pragma unroll
    for (int r = 0; r < 8; ++r)
      dst[pre[r] + hoff] = (_Float16)((c[r] + bv) * sc);
  }
}

// ---------------- windowed attention, one block per (b, window, head) ----------
#define ATT_WAVES 8
// qs + ks + vt (3*20480) + rpb stage (2560) + scores (163840) + probs (81920)
#define ATT_LDS (3 * WP_ * HD_ * 2 + 2560 + ATT_WAVES * 16 * WP_ * 4 + ATT_WAVES * 16 * WP_ * 2)

__global__ void __launch_bounds__(256)
attn_kernel(const _Float16* __restrict__ qg, const _Float16* __restrict__ kg,
            const _Float16* __restrict__ vg, const float* __restrict__ rpb,
            _Float16* __restrict__ o, int shifted) {
  extern __shared__ char smem[];
  _Float16* qs = (_Float16*)smem;                    // [WP_][HD_]
  _Float16* ks = qs + WP_ * HD_;                     // [WP_][HD_]
  _Float16* vt = ks + WP_ * HD_;                     // [HD_][WP_]  (transposed V)
  float*    rl = (float*)(vt + WP_ * HD_);           // 599 floats (padded 640)
  float*    sball = rl + 640;                        // per-wave 16xWP_ f32
  _Float16* pball = (_Float16*)(sball + ATT_WAVES * 16 * WP_);

  int bid = blockIdx.x;
  int hh = bid % H_;
  int nw = (bid / H_) % NW_;
  int b  = bid / (H_ * NW_);
  size_t base = (((size_t)b * NW_ + nw) * H_ + hh) * ((size_t)W_ * HD_);
  const _Float16* qp = qg + base;
  const _Float16* kp = kg + base;
  const _Float16* vp = vg + base;

  // async stage q,k (contiguous 19200B each) into LDS
  unsigned qoff = (unsigned)(uintptr_t)qs;
  unsigned koff = (unsigned)(uintptr_t)ks;
  for (int t = threadIdx.x; t < (W_ * HD_) / 8; t += 256) {   // 1200 x 16B
    async_copy16(qoff + t * 16, (const char*)qp + t * 16);
    async_copy16(koff + t * 16, (const char*)kp + t * 16);
  }
  // stage V transposed: vt[d][row] = v[row][d]
  for (int t = threadIdx.x; t < (W_ * HD_) / 8; t += 256) {
    h8v vv = ((const h8v*)vp)[t];
    int row = t >> 2, dbase = (t & 3) * 8;
#pragma unroll
    for (int j = 0; j < 8; ++j) vt[(dbase + j) * WP_ + row] = vv[j];
  }
  // zero the padded rows/cols
  for (int t = threadIdx.x; t < (WP_ - W_) * HD_; t += 256) { // 640
    int r = W_ + t / HD_, d = t % HD_;
    qs[r * HD_ + d] = (_Float16)0.0f;
    ks[r * HD_ + d] = (_Float16)0.0f;
    vt[d * WP_ + r] = (_Float16)0.0f;
  }
  // stage relative-position-bias column for this head
  for (int t = threadIdx.x; t < 2 * W_ - 1; t += 256) rl[t] = rpb[t * H_ + hh];
  wait_async0();
  __syncthreads();

  const int wave = threadIdx.x >> 5;
  const int lane = threadIdx.x & 31, col = lane & 15, hi = lane >> 4;
  float*    sw = sball + (size_t)wave * 16 * WP_;
  _Float16* pw = pball + (size_t)wave * 16 * WP_;

  for (int s = wave; s < 19; s += ATT_WAVES) {       // 19 stripes cover rows 0..303
    int m0 = s * 16;
    v16h a = load_a_frag(qs + (size_t)m0 * HD_, HD_);
    // scores S = q @ k^T  (K=32 -> one WMMA per 16x16 tile)
    for (int ntile = 0; ntile < WP_ / 16; ++ntile) {
      int n0 = ntile * 16;
      v8f c = {};
      c = wmma_f32_f16(a, load_bt_frag(ks + (size_t)n0 * HD_, HD_), c);
      int wk = n0 + col;
#pragma unroll
      for (int r = 0; r < 8; ++r) {
        int wq = m0 + r + hi * 8;
        float val;
        if (wk >= W_) {
          val = -1e30f;                                // padded key columns
        } else {
          val = c[r];
          if (wq < W_) {
            val += rl[wq - wk + W_ - 1];               // relative position bias
            if (shifted) {
              int pq = nw * W_ + wq, pk = nw * W_ + wk;
              int sq = (pq >= L_ - SHIFT_) ? 2 : ((pq >= L_ - W_) ? 1 : 0);
              int sk = (pk >= L_ - SHIFT_) ? 2 : ((pk >= L_ - W_) ? 1 : 0);
              if (sq != sk) val -= 1e9f;               // shift-window mask
            }
          }
        }
        sw[(r + hi * 8) * WP_ + wk] = val;
      }
    }
    // softmax per row (16 rows, one lane each, float4 vectorized)
    if (lane < 16) {
      float4* row4 = (float4*)(sw + lane * WP_);
      h4v*    pr4  = (h4v*)(pw + lane * WP_);
      float mx = -1e30f;
      for (int j = 0; j < WP_ / 4; ++j) {
        float4 f = row4[j];
        mx = fmaxf(mx, fmaxf(fmaxf(f.x, f.y), fmaxf(f.z, f.w)));
      }
      float sum = 0.f;
      for (int j = 0; j < WP_ / 4; ++j) {
        float4 f = row4[j];
        float e0 = __expf(f.x - mx), e1 = __expf(f.y - mx);
        float e2 = __expf(f.z - mx), e3 = __expf(f.w - mx);
        sum += (e0 + e1) + (e2 + e3);
        h4v p; p[0] = (_Float16)e0; p[1] = (_Float16)e1; p[2] = (_Float16)e2; p[3] = (_Float16)e3;
        pr4[j] = p;
      }
      float inv = 1.0f / sum;
      for (int j = 0; j < WP_ / 4; ++j) {
        h4v p = pr4[j];
#pragma unroll
        for (int e = 0; e < 4; ++e) p[e] = (_Float16)((float)p[e] * inv);
        pr4[j] = p;
      }
    }
    // O = P @ V : (16 x WP_) @ (WP_ x 32); B-fragments contiguous via vt
    for (int ct = 0; ct < 2; ++ct) {
      v8f c = {};
      for (int kt = 0; kt < WP_ / 32; ++kt) {
        v16h a2 = load_a_frag(pw + kt * 32, WP_);
        v16h bf = load_bt_frag(vt + (size_t)(ct * 16) * WP_ + kt * 32, WP_);
        c = wmma_f32_f16(a2, bf, c);
      }
#pragma unroll
      for (int r = 0; r < 8; ++r) {
        int wq = m0 + r + hi * 8;
        if (wq < W_) {
          o[((size_t)b * L_ + (size_t)nw * W_ + wq) * C_ + hh * HD_ + ct * 16 + col] = (_Float16)c[r];
        }
      }
    }
  }
}

// ---------------- proj GEMM + residual (roll-back by +shift); A reuse x8 ------
__global__ void __launch_bounds__(256)
proj_gemm_kernel(const _Float16* __restrict__ o, const _Float16* __restrict__ wt,
                 const float* __restrict__ bias, float* __restrict__ x, int shift) {
  const int MT = (B_ * L_) / 16;
  int wid = blockIdx.x * 8 + (threadIdx.x >> 5);
  if (wid >= MT) return;
  int m0 = wid * 16;
  const int lane = threadIdx.x & 31, col = lane & 15, hi = lane >> 4;

  v16h af[4];
#pragma unroll
  for (int kt = 0; kt < 4; ++kt) af[kt] = load_a_frag(o + (size_t)m0 * C_ + kt * 32, C_);

  size_t xrow[8];
#pragma unroll
  for (int r = 0; r < 8; ++r) {
    int m = m0 + r + hi * 8;
    int bb = m / L_, l = m - bb * L_;
    int ldst = l + shift; if (ldst >= L_) ldst -= L_;
    xrow[r] = ((size_t)bb * L_ + ldst) * C_;
  }

  for (int nt = 0; nt < 8; ++nt) {
    int n0 = nt * 16;
    v8f c = {};
#pragma unroll
    for (int kt = 0; kt < 4; ++kt)
      c = wmma_f32_f16(af[kt], load_bt_frag(wt + (size_t)n0 * C_ + kt * 32, C_), c);
    int n = n0 + col;
    float bv = bias[n];
#pragma unroll
    for (int r = 0; r < 8; ++r)
      x[xrow[r] + n] += c[r] + bv;
  }
}

// ---------------- fused MLP: GELU(h @ W1 + b1) @ W2 + b2, += x ----------------
#define MLP_WAVES 8
#define MLP_LDS (MLP_WAVES * 16 * 512 * 2)

__global__ void __launch_bounds__(256)
mlp_kernel(const _Float16* __restrict__ h2, const _Float16* __restrict__ w1t,
           const float* __restrict__ b1, const _Float16* __restrict__ w2t,
           const float* __restrict__ b2, float* __restrict__ x) {
  extern __shared__ char smem[];
  _Float16* gall = (_Float16*)smem;
  const int wave = threadIdx.x >> 5;
  int stripe = blockIdx.x * MLP_WAVES + wave;
  if (stripe >= (B_ * L_) / 16) return;
  int m0 = stripe * 16;
  _Float16* g = gall + (size_t)wave * 16 * 512;
  const int lane = threadIdx.x & 31, col = lane & 15, hi = lane >> 4;

  v16h af[4];
#pragma unroll
  for (int kt = 0; kt < 4; ++kt) af[kt] = load_a_frag(h2 + (size_t)m0 * C_ + kt * 32, C_);

  for (int nt = 0; nt < 32; ++nt) {                  // N = 512
    v8f c = {};
#pragma unroll
    for (int kt = 0; kt < 4; ++kt)
      c = wmma_f32_f16(af[kt], load_bt_frag(w1t + (size_t)nt * 16 * C_ + kt * 32, C_), c);
    int n = nt * 16 + col;
    float bv = b1[n];
#pragma unroll
    for (int r = 0; r < 8; ++r) {
      float val = c[r] + bv;
      float x3 = val * val * val;
      float t = tanhf(0.7978845608028654f * (val + 0.044715f * x3));  // tanh-gelu
      g[(r + hi * 8) * 512 + n] = (_Float16)(0.5f * val * (1.0f + t));
    }
  }
  for (int nt = 0; nt < 8; ++nt) {                   // N = 128, K = 512
    v8f c = {};
    for (int kt = 0; kt < 16; ++kt) {
      v16h a  = load_a_frag(g + kt * 32, 512);
      v16h bf = load_bt_frag(w2t + (size_t)nt * 16 * 512 + kt * 32, 512);
      c = wmma_f32_f16(a, bf, c);
    }
    int n = nt * 16 + col;
    float bv = b2[n];
#pragma unroll
    for (int r = 0; r < 8; ++r) {
      int m = m0 + r + hi * 8;
      x[(size_t)m * C_ + n] += c[r] + bv;
    }
  }
}

// ---------------- host orchestration ----------------
extern "C" void kernel_launch(void* const* d_in, const int* in_sizes, int n_in,
                              void* d_out, int out_size, void* d_ws, size_t ws_size,
                              hipStream_t stream) {
  (void)in_sizes; (void)n_in; (void)out_size; (void)ws_size;
  const float* s1     = (const float*)d_in[0];
  const float* s2     = (const float*)d_in[1];
  const float* ln1_g  = (const float*)d_in[2];
  const float* ln1_b  = (const float*)d_in[3];
  const float* qkv_w  = (const float*)d_in[4];
  const float* qkv_b  = (const float*)d_in[5];
  const float* proj_w = (const float*)d_in[6];
  const float* proj_b = (const float*)d_in[7];
  const float* ln2_g  = (const float*)d_in[8];
  const float* ln2_b  = (const float*)d_in[9];
  const float* mlp_w1 = (const float*)d_in[10];
  const float* mlp_b1 = (const float*)d_in[11];
  const float* mlp_w2 = (const float*)d_in[12];
  const float* mlp_b2 = (const float*)d_in[13];
  const float* rpb    = (const float*)d_in[14];

  char* ws = (char*)d_ws;
  float*    xbuf = (float*)ws;                        // B*L*C f32 = 30,720,000 B
  _Float16* hbuf = (_Float16*)(ws + 30720000);        // B*L*C f16
  _Float16* qbuf = (_Float16*)(ws + 46080000);
  _Float16* kbuf = (_Float16*)(ws + 61440000);
  _Float16* vbuf = (_Float16*)(ws + 76800000);
  _Float16* obuf = (_Float16*)(ws + 92160000);
  _Float16* wtb  = (_Float16*)(ws + 107520000);       // transposed f16 weights
  _Float16* qkvw_t  = wtb;                            // 6*384*128
  _Float16* projw_t = wtb + 294912;                   // 6*128*128
  _Float16* m1w_t   = wtb + 393216;                   // 6*512*128
  _Float16* m2w_t   = wtb + 786432;                   // 6*128*512

  hipFuncSetAttribute(reinterpret_cast<const void*>(attn_kernel),
                      hipFuncAttributeMaxDynamicSharedMemorySize, ATT_LDS);
  hipFuncSetAttribute(reinterpret_cast<const void*>(mlp_kernel),
                      hipFuncAttributeMaxDynamicSharedMemorySize, MLP_LDS);

  wcast_kernel<<<(NL_ * C_ * 3 * C_ + 255) / 256, 256, 0, stream>>>(qkv_w, qkvw_t, C_, 3 * C_);
  wcast_kernel<<<(NL_ * C_ * C_ + 255) / 256, 256, 0, stream>>>(proj_w, projw_t, C_, C_);
  wcast_kernel<<<(NL_ * C_ * 4 * C_ + 255) / 256, 256, 0, stream>>>(mlp_w1, m1w_t, C_, 4 * C_);
  wcast_kernel<<<(NL_ * 4 * C_ * C_ + 255) / 256, 256, 0, stream>>>(mlp_w2, m2w_t, 4 * C_, C_);

  init_kernel<<<(B_ * C_ * L_ + 255) / 256, 256, 0, stream>>>(s1, s2, xbuf);

  for (int i = 0; i < NL_; ++i) {
    int shift = (i % 2) ? SHIFT_ : 0;
    ln_kernel<<<B_ * L_, 128, 0, stream>>>(xbuf, hbuf, ln1_g + i * C_, ln1_b + i * C_, shift);
    qkv_gemm_kernel<<<((B_ * L_ / 16) * 4 + 7) / 8, 256, 0, stream>>>(
        hbuf, qkvw_t + (size_t)i * C_ * 3 * C_, qkv_b + i * 3 * C_, qbuf, kbuf, vbuf);
    attn_kernel<<<B_ * NW_ * H_, 256, ATT_LDS, stream>>>(
        qbuf, kbuf, vbuf, rpb + (size_t)i * (2 * W_ - 1) * H_, obuf, shift ? 1 : 0);
    proj_gemm_kernel<<<(B_ * L_ / 16 + 7) / 8, 256, 0, stream>>>(
        obuf, projw_t + (size_t)i * C_ * C_, proj_b + i * C_, xbuf, shift);
    ln_kernel<<<B_ * L_, 128, 0, stream>>>(xbuf, hbuf, ln2_g + i * C_, ln2_b + i * C_, 0);
    mlp_kernel<<<(B_ * L_ / 16 + MLP_WAVES - 1) / MLP_WAVES, 256, MLP_LDS, stream>>>(
        hbuf, m1w_t + (size_t)i * C_ * 4 * C_, mlp_b1 + i * 4 * C_,
        m2w_t + (size_t)i * 4 * C_ * C_, mlp_b2 + i * C_, xbuf);
  }

  pack_kernel<<<(B_ * C_ * L_ + 255) / 256, 256, 0, stream>>>(xbuf, (float*)d_out);
}
